// MyQuantizedConv2d_60498909331539
// MI455X (gfx1250) — 8-bit matrix operand order repaired and checked by reading, compiled, not run on hardware
//
#include <hip/hip_runtime.h>
#include <hip/hip_bf16.h>

typedef int v8i __attribute__((ext_vector_type(8)));

#define NB    32
#define CIN   128
#define COUT  256
#define HH    56
#define WW    56
#define HW    3136
#define KTOT  1152
#define NCHUNK 18
#define I32_MIN (-2147483647-1)
#define I32_MAX 2147483647

#define WS_X8   0
#define WS_S    12845056
#define WS_AP   13246464
#define WS_B2   13541376

__global__ void k_pack_input(const int* __restrict__ in, signed char* __restrict__ x8, int n) {
    int i = blockIdx.x * blockDim.x + threadIdx.x;
    if (i < n) x8[i] = (signed char)in[i];
}

__global__ void k_chsum(const signed char* __restrict__ x8, int* __restrict__ S) {
    int i = blockIdx.x * blockDim.x + threadIdx.x;
    if (i >= NB * HW) return;
    int n = i / HW, p = i % HW;
    const signed char* base = x8 + (size_t)n * CIN * HW + p;
    int s = 0;
#pragma unroll 8
    for (int ci = 0; ci < CIN; ++ci) s += base[ci * HW];
    S[i] = s;
}

__global__ void k_bias2(const int* __restrict__ qw, const int* __restrict__ qbias,
                        const int* __restrict__ pzpI, const int* __restrict__ pzpW,
                        int* __restrict__ bias2) {
    int co = blockIdx.x * blockDim.x + threadIdx.x;
    if (co >= COUT) return;
    int izp = pzpI[0], wzp = pzpW[0];
    const int* b = qw + (size_t)co * (CIN * 9);
    int s = 0;
    for (int j = 0; j < CIN * 9; ++j) s += b[j];
    bias2[co] = qbias[co] - izp * s + KTOT * izp * wzp;
}

__global__ void k_pack_w(const int* __restrict__ qw, int* __restrict__ Ap) {
    int blk = blockIdx.x;
    int mt = blk / NCHUNK, ch = blk % NCHUNK;
    int lane = threadIdx.x & 31;
    int hi = lane >> 4, m = lane & 15;
    int co = mt * 16 + m;
    int* dst = Ap + (((size_t)mt * NCHUNK + ch) * 32 + lane) * 8;
#pragma unroll
    for (int v = 0; v < 8; ++v) {
        unsigned d = 0;
#pragma unroll
        for (int b = 0; b < 4; ++b) {
            int kl = ((v >> 2) << 5) + (hi << 4) + ((v & 3) << 2) + b;
            int kg = ch * 64 + kl;
            int tap = kg >> 7;
            int ci = kg & 127;
            int wv = qw[((size_t)co * CIN + ci) * 9 + tap];
            d |= ((unsigned)(wv & 0xff)) << (8 * b);
        }
        dst[v] = (int)d;
    }
}

__global__ __launch_bounds__(256)
void k_conv_gemm(const signed char* __restrict__ x8,
                 const int* __restrict__ S,
                 const int* __restrict__ Ap,
                 const int* __restrict__ bias2,
                 float* __restrict__ out,
                 const int* __restrict__ pzpI, const int* __restrict__ pzpW,
                 const int* __restrict__ pzpO, const int* __restrict__ pqm,
                 const int* __restrict__ prs) {
    __shared__ __align__(32) int ldsB[2][4][32][8];
    __shared__ int ldsPS[64];

    const int t = threadIdx.x;
    const int lane = t & 31;
    const int wave = t >> 5;
    const int pblock = blockIdx.x;
    const int mblock = blockIdx.y;
    const int n = blockIdx.z;

    const int izp = pzpI[0], wzp = pzpW[0], ozp = pzpO[0];
    const int qmul = pqm[0], rsh = prs[0];

    const int c  = t & 63;
    const int kq = t >> 6;
    const int p  = pblock * 64 + c;
    const int h  = p / WW, w = p % WW;
    const int ntile = c >> 4;
    const int laneL = (c & 15) | ((kq & 1) << 4);
    const int vbase = (kq >> 1) << 2;
    const unsigned izp4 = ((unsigned)izp & 0xffu) * 0x01010101u;
    const signed char* xbase = x8 + ((size_t)n * CIN + (kq << 4)) * HW;

    auto fill = [&](int ch) {
        int tap = ch >> 1;
        int dy = tap / 3 - 1, dx = tap % 3 - 1;
        int y = h + dy, x = w + dx;
        bool valid = ((unsigned)y < HH) && ((unsigned)x < WW);
        int off = valid ? (y * WW + x) : (h * WW + w);
        const signed char* src = xbase + (size_t)((ch & 1) << 6) * HW + off;
        int d[4];
#pragma unroll
        for (int g = 0; g < 4; ++g) {
            unsigned dw = 0;
#pragma unroll
            for (int b = 0; b < 4; ++b)
                dw |= ((unsigned)(unsigned char)src[(g * 4 + b) * HW]) << (8 * b);
            d[g] = valid ? (int)dw : (int)izp4;
        }
        int* dst = &ldsB[ch & 1][ntile][laneL][vbase];
        *(int4*)dst = make_int4(d[0], d[1], d[2], d[3]);
    };

    if (t < 64) {
        int s = 0;
#pragma unroll
        for (int dy = -1; dy <= 1; ++dy)
#pragma unroll
            for (int dx = -1; dx <= 1; ++dx) {
                int y = h + dy, x = w + dx;
                s += (((unsigned)y < HH) && ((unsigned)x < WW))
                         ? S[n * HW + y * WW + x] : izp * CIN;
            }
        ldsPS[c] = -wzp * s;
    }

    v8i acc[4];
#pragma unroll
    for (int i = 0; i < 4; ++i) acc[i] = (v8i){0, 0, 0, 0, 0, 0, 0, 0};

    const int mt = mblock * 8 + wave;
    const int* Abase = Ap + (size_t)mt * NCHUNK * 256 + lane * 8;

    fill(0);
    __syncthreads();

#pragma unroll
    for (int ch = 0; ch < NCHUNK; ++ch) {
        v8i a = *(const v8i*)(Abase + ch * 256);
        if (ch + 1 < NCHUNK)
            __builtin_prefetch(Abase + (ch + 1) * 256, 0, 3);
#pragma unroll
        for (int nt = 0; nt < 4; ++nt) {
            v8i b = *(const v8i*)(&ldsB[ch & 1][nt][lane][0]);
            acc[nt] = __builtin_amdgcn_wmma_i32_16x16x64_iu8(
                true, a, true, b, acc[nt], false, false);
        }
        if (ch + 1 < NCHUNK) fill(ch + 1);
        __syncthreads();
    }

    const int coBase = mt * 16 + ((lane >> 4) << 3);
    const int colBase = lane & 15;
    const v8i bz = *(const v8i*)(bias2 + coBase);
    float* outBase = out + (long long)n * COUT * HW
                         + (long long)coBase * HW + pblock * 64;
#pragma unroll
    for (int nt = 0; nt < 4; ++nt) {
        int col = nt * 16 + colBase;
        int psN = ldsPS[col];
        float* o0 = outBase + col;
#pragma unroll
        for (int i = 0; i < 8; ++i) {
            int z = acc[nt][i] + bz[i] + psN;
            long long ab = (long long)z * (long long)qmul;
            long long nudge = (ab >= 0) ? (1ll << 30) : (1ll - (1ll << 30));
            long long q = ab + nudge;
            int res = (int)((q + ((q >> 63) & 0x7fffffffll)) >> 31);
            if (z == I32_MIN && qmul == I32_MIN) res = I32_MAX;
            int mask = (1 << rsh) - 1;
            int rem = res & mask;
            int thr = (mask >> 1) + ((res < 0) ? 1 : 0);
            int o = (res >> rsh) + ((rem > thr) ? 1 : 0);
            o += ozp;
            o = o < -127 ? -127 : (o > 127 ? 127 : o);
            o0[(long long)i * HW] = (float)o;
        }
    }
}

extern "C" void kernel_launch(void* const* d_in, const int* in_sizes, int n_in,
                              void* d_out, int out_size, void* d_ws, size_t ws_size,
                              hipStream_t stream) {
    const int* in   = (const int*)d_in[0];
    const int* qw   = (const int*)d_in[1];
    const int* qb   = (const int*)d_in[2];
    const int* zpI  = (const int*)d_in[3];
    const int* zpW  = (const int*)d_in[4];
    const int* zpO  = (const int*)d_in[5];
    const int* qm   = (const int*)d_in[6];
    const int* rs   = (const int*)d_in[7];
    float* out = (float*)d_out;

    char* ws = (char*)d_ws;
    signed char* x8 = (signed char*)(ws + WS_X8);
    int* S     = (int*)(ws + WS_S);
    int* Ap    = (int*)(ws + WS_AP);
    int* bias2 = (int*)(ws + WS_B2);

    const int nelem = NB * CIN * HW;
    k_pack_input<<<(nelem + 255) / 256, 256, 0, stream>>>(in, x8, nelem);
    k_chsum<<<(NB * HW + 255) / 256, 256, 0, stream>>>(x8, S);
    k_pack_w<<<16 * NCHUNK, 32, 0, stream>>>(qw, Ap);
    k_bias2<<<1, 256, 0, stream>>>(qw, qb, zpI, zpW, bias2);

    dim3 grid(49, 2, NB);
    k_conv_gemm<<<grid, 256, 0, stream>>>(x8, S, Ap, bias2, out,
                                          zpI, zpW, zpO, qm, rs);
}
